// ThmNet_24068996726972
// MI455X (gfx1250) — matrix operand order repaired and checked by reading, not compiled, not run on hardware
//
#include <hip/hip_runtime.h>
#include <hip/hip_bf16.h>
#include <math.h>

// Problem constants (match reference)
#define NND 131072
#define NE  131072
#define HH  128
#define GG  1024
#define BB  128
#define NLEM 2000

typedef __bf16 bf16;
typedef __attribute__((ext_vector_type(16))) __bf16 v16bf;
typedef __attribute__((ext_vector_type(8)))  __bf16 v8bf;
typedef __attribute__((ext_vector_type(4)))  __bf16 v4bf;
typedef __attribute__((ext_vector_type(8)))  float  v8f;

union AFrag { v16bf v; v8bf h[2]; bf16 e[16]; };
union CFrag { v8f   v; float e[8];  };
union Pk4   { v4bf  v; bf16 e[4];   };

__device__ __forceinline__ float sigmoidf_(float x) { return 1.0f / (1.0f + __expf(-x)); }

// LDS tiles: row stride 40 bf16 = 80 B (16B-aligned rows, staggered banks)
#define AS_LD 40
#define BS_LD 40

// Fragments from staged LDS tiles (all contiguous 16B runs -> ds_load_b128):
// A (16-bit 16x32 layout): lane row = nlane;
//   e[0..7]  = As[row][ 8*half .. +7],  e[8..15] = As[row][16+8*half .. +7]
// B staged transposed (BsT[n][k]): lane col = nlane;
//   e[0..7]  = BsT[col][ 8*half .. +7],  e[8..15] = BsT[col][16+8*half .. +7]
__device__ __forceinline__ void load_frags(const bf16* __restrict__ arow,
                                           const bf16* __restrict__ brow,
                                           int half, AFrag& fa, AFrag& fb)
{
    fa.h[0] = *(const v8bf*)(arow + 8 * half);
    fa.h[1] = *(const v8bf*)(arow + 16 + 8 * half);
    fb.h[0] = *(const v8bf*)(brow + 8 * half);
    fb.h[1] = *(const v8bf*)(brow + 16 + 8 * half);
}

// ---------------------------------------------------------------------------
// Generic GEMM: Out[row*ldo+col] = act( A[gathA(row),:K] @ B[:,col] + bias[col]
//                                       + AddM[gathM(row)*ldm+col] )
// bf16 WMMA 16x16x32, f32 accumulate. Block = 256 thr = 8 waves = 32x64 tile.
// Software-pipelined: chunk i+1 global loads issued before chunk i's WMMA.
// act: 0=none 1=relu 2=sigmoid
// ---------------------------------------------------------------------------
__global__ __launch_bounds__(256)
void gemm_wmma(const float* __restrict__ A, const int* __restrict__ rowIdxA, int lda,
               const float* __restrict__ Bm, int Nc, int K,
               const float* __restrict__ bias,
               const float* __restrict__ AddM, const int* __restrict__ rowIdxM, int ldm,
               float* __restrict__ Out, int ldo, int M, int act)
{
    __shared__ bf16 As[32][AS_LD];
    __shared__ bf16 BsT[64][BS_LD];

    const int tid   = threadIdx.x;
    const int lane  = tid & 31;
    const int wave  = tid >> 5;
    const int wm    = wave >> 2;   // 0..1 (M sub-tile)
    const int wn    = wave & 3;    // 0..3 (N sub-tile)
    const int m0    = blockIdx.x * 32;
    const int n0    = blockIdx.y * 64;
    const int half  = lane >> 4;
    const int nlane = lane & 15;

    // ---- branchless staging setup ----
    // A: one float4 (4 k) of one row per thread
    const int rA    = tid >> 3;    // 0..31
    const int kqA   = tid & 7;     // k = kqA*4 .. +3
    const int growA = m0 + rA;
    const int rclA  = (growA < M) ? growA : (M - 1);
    const float mulA = (growA < M) ? 1.0f : 0.0f;
    const int gaA   = rowIdxA ? rowIdxA[rclA] : rclA;
    const float* aptr = A + (size_t)gaA * lda + kqA * 4;

    // B: two columns per thread, 4 consecutive k each
    const int colB0 = tid >> 3;    // 0..31 (second col = +32)
    const int kqB   = tid & 7;
    int   gcol0 = n0 + colB0;
    int   gcol1 = n0 + colB0 + 32;
    const float mulB0 = (gcol0 < Nc) ? 1.0f : 0.0f;
    const float mulB1 = (gcol1 < Nc) ? 1.0f : 0.0f;
    gcol0 = (gcol0 < Nc) ? gcol0 : (Nc - 1);
    gcol1 = (gcol1 < Nc) ? gcol1 : (Nc - 1);
    const float* bptr0 = Bm + (size_t)(kqB * 4) * Nc + gcol0;
    const float* bptr1 = Bm + (size_t)(kqB * 4) * Nc + gcol1;

    CFrag acc;
#pragma unroll
    for (int i = 0; i < 8; ++i) acc.e[i] = 0.0f;

    const bf16* arow = &As[wm * 16 + nlane][0];
    const bf16* brow = &BsT[wn * 16 + nlane][0];

    // ---- pipeline prologue: raw loads for chunk 0 ----
    float4 ra = *(const float4*)(aptr);
    float  rb0[4], rb1[4];
#pragma unroll
    for (int j = 0; j < 4; ++j) {
        rb0[j] = bptr0[(size_t)j * Nc];
        rb1[j] = bptr1[(size_t)j * Nc];
    }

    for (int k0 = 0; k0 < K; k0 += 32) {
        // convert chunk-i raw registers to bf16 and stage into LDS
        Pk4 pa, pb0, pb1;
        pa.e[0] = (bf16)(ra.x * mulA); pa.e[1] = (bf16)(ra.y * mulA);
        pa.e[2] = (bf16)(ra.z * mulA); pa.e[3] = (bf16)(ra.w * mulA);
#pragma unroll
        for (int j = 0; j < 4; ++j) {
            pb0.e[j] = (bf16)(rb0[j] * mulB0);
            pb1.e[j] = (bf16)(rb1[j] * mulB1);
        }
        *(v4bf*)&As[rA][kqA * 4]          = pa.v;
        *(v4bf*)&BsT[colB0][kqB * 4]      = pb0.v;
        *(v4bf*)&BsT[colB0 + 32][kqB * 4] = pb1.v;
        __syncthreads();

        // issue chunk-(i+1) global loads (clamped: last iter reloads same chunk)
        const int kn = (k0 + 32 < K) ? (k0 + 32) : k0;
        ra = *(const float4*)(aptr + kn);
        const float* bn0 = bptr0 + (size_t)kn * Nc;
        const float* bn1 = bptr1 + (size_t)kn * Nc;
#pragma unroll
        for (int j = 0; j < 4; ++j) {
            rb0[j] = bn0[(size_t)j * Nc];
            rb1[j] = bn1[(size_t)j * Nc];
        }

        // fragments + matrix op for chunk i
        AFrag fa, fb;
        load_frags(arow, brow, half, fa, fb);
        acc.v = __builtin_amdgcn_wmma_f32_16x16x32_bf16(
            false, fa.v, false, fb.v, (short)0, acc.v, false, false);
        __syncthreads();
    }

    // Epilogue: elem r -> row (r + 8*half), col fixed per lane
    const int col = n0 + wn * 16 + nlane;
    const float bcol = (bias && col < Nc) ? bias[col] : 0.0f;
#pragma unroll
    for (int r = 0; r < 8; ++r) {
        const int grow = m0 + wm * 16 + r + 8 * half;
        if (grow < M && col < Nc) {
            float v = acc.e[r] + bcol;
            if (AddM) {
                int gm = rowIdxM ? rowIdxM[grow] : grow;
                v += AddM[(size_t)gm * ldm + col];
            }
            if (act == 1)      v = fmaxf(v, 0.0f);
            else if (act == 2) v = sigmoidf_(v);
            Out[(size_t)grow * ldo + col] = v;
        }
    }
}

// ---------------------------------------------------------------------------
// Per-edge forget gate + fc scatter, fused:
//   fe = sigmoid( h[src[e]] @ U_f + x_f[dst[e]] )    (WMMA over K=128)
//   fc[dst[e]] += fe * c[src[e]]                     (f32 atomics)
// ---------------------------------------------------------------------------
__global__ __launch_bounds__(256)
void edge_forget_wmma(const float* __restrict__ h, const float* __restrict__ cst,
                      const float* __restrict__ Uf, const float* __restrict__ xf,
                      const int* __restrict__ sidx, const int* __restrict__ didx,
                      float* __restrict__ fc, int E)
{
    __shared__ bf16 As[32][AS_LD];
    __shared__ bf16 BsT[64][BS_LD];

    const int tid   = threadIdx.x;
    const int lane  = tid & 31;
    const int wave  = tid >> 5;
    const int wm    = wave >> 2;
    const int wn    = wave & 3;
    const int m0    = blockIdx.x * 32;   // edge tile
    const int n0    = blockIdx.y * 64;   // feature tile (HH=128 -> gridDim.y=2)
    const int half  = lane >> 4;
    const int nlane = lane & 15;

    const int rA   = tid >> 3;
    const int kqA  = tid & 7;
    const int geA  = m0 + rA;
    const int eclA = (geA < E) ? geA : (E - 1);
    const float mulA = (geA < E) ? 1.0f : 0.0f;
    const int sA   = sidx[eclA];
    const float* aptr = h + (size_t)sA * HH + kqA * 4;

    const int colB0 = tid >> 3;
    const int kqB   = tid & 7;
    const float* bptr0 = Uf + (size_t)(kqB * 4) * HH + n0 + colB0;
    const float* bptr1 = bptr0 + 32;

    CFrag acc;
#pragma unroll
    for (int i = 0; i < 8; ++i) acc.e[i] = 0.0f;

    const bf16* arow = &As[wm * 16 + nlane][0];
    const bf16* brow = &BsT[wn * 16 + nlane][0];

    float4 ra = *(const float4*)(aptr);
    float  rb0[4], rb1[4];
#pragma unroll
    for (int j = 0; j < 4; ++j) {
        rb0[j] = bptr0[j * HH];
        rb1[j] = bptr1[j * HH];
    }

    for (int k0 = 0; k0 < HH; k0 += 32) {
        Pk4 pa, pb0, pb1;
        pa.e[0] = (bf16)(ra.x * mulA); pa.e[1] = (bf16)(ra.y * mulA);
        pa.e[2] = (bf16)(ra.z * mulA); pa.e[3] = (bf16)(ra.w * mulA);
#pragma unroll
        for (int j = 0; j < 4; ++j) {
            pb0.e[j] = (bf16)rb0[j];
            pb1.e[j] = (bf16)rb1[j];
        }
        *(v4bf*)&As[rA][kqA * 4]          = pa.v;
        *(v4bf*)&BsT[colB0][kqB * 4]      = pb0.v;
        *(v4bf*)&BsT[colB0 + 32][kqB * 4] = pb1.v;
        __syncthreads();

        const int kn = (k0 + 32 < HH) ? (k0 + 32) : k0;
        ra = *(const float4*)(aptr + kn);
        const float* bn0 = bptr0 + (size_t)kn * HH;
        const float* bn1 = bptr1 + (size_t)kn * HH;
#pragma unroll
        for (int j = 0; j < 4; ++j) {
            rb0[j] = bn0[j * HH];
            rb1[j] = bn1[j * HH];
        }

        AFrag fa, fb;
        load_frags(arow, brow, half, fa, fb);
        acc.v = __builtin_amdgcn_wmma_f32_16x16x32_bf16(
            false, fa.v, false, fb.v, (short)0, acc.v, false, false);
        __syncthreads();
    }

    const int col = n0 + wn * 16 + nlane;
#pragma unroll
    for (int r = 0; r < 8; ++r) {
        const int ge = m0 + wm * 16 + r + 8 * half;
        if (ge < E) {
            int s = sidx[ge], d = didx[ge];
            float fe = sigmoidf_(acc.e[r] + xf[(size_t)d * HH + col]);
            atomicAdd(&fc[(size_t)d * HH + col], fe * cst[(size_t)s * HH + col]);
        }
    }
}

// hs[dst[e], :] += h[src[e], :]
__global__ void hs_scatter(const float* __restrict__ h, const int* __restrict__ s,
                           const int* __restrict__ d, float* __restrict__ hs, int E)
{
    size_t idx = (size_t)blockIdx.x * blockDim.x + threadIdx.x;
    if (idx >= (size_t)E * HH) return;
    int e = (int)(idx >> 7), j = (int)(idx & 127);
    atomicAdd(&hs[(size_t)d[e] * HH + j], h[(size_t)s[e] * HH + j]);
}

// Gates: i,o sigmoid, u tanh; c2 = i*u + fc; h2 = o*tanh(c2)
__global__ void node_update(const float* __restrict__ iou, const float* __restrict__ fc,
                            float* __restrict__ h, float* __restrict__ c, int n)
{
    size_t idx = (size_t)blockIdx.x * blockDim.x + threadIdx.x;
    if (idx >= (size_t)n * HH) return;
    int nd = (int)(idx >> 7), j = (int)(idx & 127);
    const float* row = iou + (size_t)nd * 384;
    float i = sigmoidf_(row[j]);
    float o = sigmoidf_(row[128 + j]);
    float u = tanhf(row[256 + j]);
    float fcv = fc ? fc[idx] : 0.0f;
    float c2 = i * u + fcv;
    c[idx] = c2;
    h[idx] = o * tanhf(c2);
}

__global__ void pool_nodes(const float* __restrict__ st, const int* __restrict__ gidx,
                           float* __restrict__ pg, int n)
{
    size_t idx = (size_t)blockIdx.x * blockDim.x + threadIdx.x;
    if (idx >= (size_t)n * HH) return;
    int nd = (int)(idx >> 7), j = (int)(idx & 127);
    atomicAdd(&pg[(size_t)gidx[nd] * HH + j], st[idx]);
}

__global__ void pool_graphs(const float* __restrict__ pg, const int* __restrict__ bidx,
                            float* __restrict__ obj, int g)
{
    size_t idx = (size_t)blockIdx.x * blockDim.x + threadIdx.x;
    if (idx >= (size_t)g * HH) return;
    int gi = (int)(idx >> 7), j = (int)(idx & 127);
    atomicAdd(&obj[(size_t)bidx[gi] * HH + j], pg[idx]);
}

__global__ void build_out256(const float* __restrict__ obj, float* __restrict__ o256)
{
    int idx = blockIdx.x * blockDim.x + threadIdx.x;
    if (idx >= BB * 256) return;
    int b = idx >> 8, cjj = idx & 255;
    o256[idx] = (cjj < 128) ? obj[b * 128 + cjj] : 0.0f;
}

// vf[b] = relu(out256[b] @ vf_w1 + vf_b1) @ vf_w2 + vf_b2 -> d_out[b*2001]
__global__ __launch_bounds__(128)
void vf_head(const float* __restrict__ o256, const float* __restrict__ w1,
             const float* __restrict__ b1, const float* __restrict__ w2,
             const float* __restrict__ b2, float* __restrict__ dout)
{
    __shared__ float red[128];
    int b = blockIdx.x, j = threadIdx.x;
    float acc = b1[j];
    for (int k = 0; k < 256; ++k) acc += o256[b * 256 + k] * w1[k * 128 + j];
    red[j] = fmaxf(acc, 0.0f) * w2[j];
    __syncthreads();
    for (int s = 64; s > 0; s >>= 1) {
        if (j < s) red[j] += red[j + s];
        __syncthreads();
    }
    if (j == 0) dout[(size_t)b * 2001] = red[0] + b2[0];
}

extern "C" void kernel_launch(void* const* d_in, const int* in_sizes, int n_in,
                              void* d_out, int out_size, void* d_ws, size_t ws_size,
                              hipStream_t stream)
{
    (void)in_sizes; (void)n_in; (void)out_size; (void)ws_size;
    const float* x        = (const float*)d_in[0];
    const int*   src      = (const int*)  d_in[1];
    const int*   dst      = (const int*)  d_in[2];
    const int*   gnn_ind  = (const int*)  d_in[3];
    const int*   bat_ind  = (const int*)  d_in[4];
    const float* W_iou    = (const float*)d_in[5];
    const float* U_iou    = (const float*)d_in[6];
    const float* b_iou    = (const float*)d_in[7];
    const float* W_f      = (const float*)d_in[8];
    const float* U_f      = (const float*)d_in[9];
    const float* b_f      = (const float*)d_in[10];
    const float* W_out    = (const float*)d_in[11];
    const float* b_out    = (const float*)d_in[12];
    const float* rb_w1    = (const float*)d_in[13];
    const float* rb_b1    = (const float*)d_in[14];
    const float* rb_w2    = (const float*)d_in[15];
    const float* rb_b2    = (const float*)d_in[16];
    const float* lq_w1    = (const float*)d_in[17];
    const float* lq_b1    = (const float*)d_in[18];
    const float* lq_w2    = (const float*)d_in[19];
    const float* lq_b2    = (const float*)d_in[20];
    const float* lo_w     = (const float*)d_in[21];
    const float* lo_b     = (const float*)d_in[22];
    const float* vf_w1    = (const float*)d_in[23];
    const float* vf_b1    = (const float*)d_in[24];
    const float* vf_w2    = (const float*)d_in[25];
    const float* vf_b2    = (const float*)d_in[26];
    float* out = (float*)d_out;

    // Workspace carve-up (floats)
    float* ws   = (float*)d_ws;
    float* xiou = ws;                              // N x 384
    float* xf   = xiou + (size_t)NND * 384;        // N x 128
    float* h    = xf   + (size_t)NND * 128;        // N x 128
    float* c    = h    + (size_t)NND * 128;        // N x 128
    float* hs   = c    + (size_t)NND * 128;        // N x 128
    float* fc   = hs   + (size_t)NND * 128;        // N x 128
    float* iou  = fc   + (size_t)NND * 128;        // N x 384
    float* st0  = iou  + (size_t)NND * 384;        // N x 128 (concat [hf|hb])
    // Reused after TreeLSTM passes:
    float* tbuf  = hs;                             // resblock hidden (N x 128)
    float* stbuf = fc;                             // state_tensor    (N x 128)
    float* pg    = iou;                            // G x 128
    float* obj   = pg   + (size_t)GG * 128;        // B x 128
    float* o256  = obj  + (size_t)BB * 128;        // B x 256
    float* t1    = o256 + (size_t)BB * 256;        // B x 256
    float* z2    = t1   + (size_t)BB * 256;        // B x 256

    const int NT32 = NND / 32;                     // 4096 row tiles
    const int ETH  = (int)(((size_t)NND * 128 + 255) / 256);
    const size_t NH_BYTES = (size_t)NND * 128 * sizeof(float);

    // Loop-invariant projections: x_iou = x@W_iou + b_iou, x_f = x@W_f + b_f
    gemm_wmma<<<dim3(NT32, 6), 256, 0, stream>>>(x, nullptr, 128, W_iou, 384, 128,
        b_iou, nullptr, nullptr, 0, xiou, 384, NND, 0);
    gemm_wmma<<<dim3(NT32, 2), 256, 0, stream>>>(x, nullptr, 128, W_f, 128, 128,
        b_f, nullptr, nullptr, 0, xf, 128, NND, 0);

    // Two TreeLSTM passes: (src->dst) then reversed (dst->src)
    for (int pass = 0; pass < 2; ++pass) {
        const int* s = (pass == 0) ? src : dst;
        const int* d = (pass == 0) ? dst : src;
        const int colOff = (pass == 0) ? 0 : 64;

        hipMemsetAsync(h, 0, NH_BYTES, stream);
        hipMemsetAsync(c, 0, NH_BYTES, stream);
        // Step 0: h=c=0 -> iou = x_iou, fc = 0 (skip both GEMMs)
        node_update<<<ETH, 256, 0, stream>>>(xiou, nullptr, h, c, NND);

        for (int step = 1; step < 3; ++step) {
            hipMemsetAsync(hs, 0, NH_BYTES, stream);
            hipMemsetAsync(fc, 0, NH_BYTES, stream);
            hs_scatter<<<ETH, 256, 0, stream>>>(h, s, d, hs, NE);
            edge_forget_wmma<<<dim3(NE / 32, 2), 256, 0, stream>>>(h, c, U_f, xf,
                s, d, fc, NE);
            gemm_wmma<<<dim3(NT32, 6), 256, 0, stream>>>(hs, nullptr, 128, U_iou,
                384, 128, nullptr, xiou, nullptr, 384, iou, 384, NND, 0);
            node_update<<<ETH, 256, 0, stream>>>(iou, fc, h, c, NND);
        }
        // h @ W_out + b_out -> st0 columns [colOff, colOff+64)
        gemm_wmma<<<dim3(NT32, 1), 256, 0, stream>>>(h, nullptr, 128, W_out, 64, 128,
            b_out, nullptr, nullptr, 0, st0 + colOff, 128, NND, 0);
    }

    // state_tensor = relu(st0 + relu(st0@rb_w1+rb_b1)@rb_w2 + rb_b2)
    gemm_wmma<<<dim3(NT32, 2), 256, 0, stream>>>(st0, nullptr, 128, rb_w1, 128, 128,
        rb_b1, nullptr, nullptr, 0, tbuf, 128, NND, 1);
    gemm_wmma<<<dim3(NT32, 2), 256, 0, stream>>>(tbuf, nullptr, 128, rb_w2, 128, 128,
        rb_b2, st0, nullptr, 128, stbuf, 128, NND, 1);

    // Two-level segment sums: node->graph->batch
    hipMemsetAsync(pg, 0, (size_t)GG * 128 * sizeof(float), stream);
    hipMemsetAsync(obj, 0, (size_t)BB * 128 * sizeof(float), stream);
    pool_nodes<<<ETH, 256, 0, stream>>>(stbuf, gnn_ind, pg, NND);
    pool_graphs<<<(GG * 128 + 255) / 256, 256, 0, stream>>>(pg, bat_ind, obj, GG);
    build_out256<<<(BB * 256 + 255) / 256, 256, 0, stream>>>(obj, o256);

    // Heads
    vf_head<<<BB, 128, 0, stream>>>(o256, vf_w1, vf_b1, vf_w2, vf_b2, out);
    gemm_wmma<<<dim3(BB / 32, 4), 256, 0, stream>>>(o256, nullptr, 256, lq_w1, 256,
        256, lq_b1, nullptr, nullptr, 0, t1, 256, BB, 1);
    gemm_wmma<<<dim3(BB / 32, 4), 256, 0, stream>>>(t1, nullptr, 256, lq_w2, 256,
        256, lq_b2, o256, nullptr, 256, z2, 256, BB, 1);
    gemm_wmma<<<dim3(BB / 32, 32), 256, 0, stream>>>(z2, nullptr, 256, lo_w, NLEM,
        256, lo_b, nullptr, nullptr, 0, out + 1, NLEM + 1, BB, 0);
}
